// ConvTranspose2d_13091060318891
// MI455X (gfx1250) — matrix operand order repaired and checked by reading, not compiled, not run on hardware
//
#include <hip/hip_runtime.h>

typedef __attribute__((ext_vector_type(16))) _Float16 v16h;
typedef __attribute__((ext_vector_type(8)))  _Float16 v8h;
typedef __attribute__((ext_vector_type(8)))  float    v8f;

#define CIN   256
#define COUT  128
#define HIN   64
#define WIN   64
#define HOUT  128
#define KTOT  1024          // Cin * 4 taps
#define NCHUNK 8            // K chunks of 128 (32 ci x 4 taps)
#define XROW_STRIDE 68      // f32 pitch of staged x rows (padded)
#define BP_STRIDE   136     // f16 pitch per pixel in B panel (bank-staggered)

// ---------------------------------------------------------------------------
// Prep: repack weight (Cin,Cout,4,4) f32 -> Ah[quad][co][k] f16,
// k = ci*4 + a*2 + b with the tap mapping per output parity class:
//   ph=0: a=0->kh=1(ihoff 0), a=1->kh=3(ihoff -1);  ph=1: a=0->kh=0(+1), a=1->kh=2(0)
// (same for pw/kw/iwoff).
// ---------------------------------------------------------------------------
__global__ void pack_weights_kernel(const float* __restrict__ w,
                                    _Float16* __restrict__ Ah) {
  int idx = blockIdx.x * 256 + threadIdx.x;        // 4*128*1024 = 524288
  if (idx >= 4 * COUT * KTOT) return;
  int k    = idx & (KTOT - 1);
  int co   = (idx >> 10) & (COUT - 1);
  int quad = idx >> 17;
  int ph = quad >> 1, pw = quad & 1;
  int ci = k >> 2;
  int a  = (k >> 1) & 1;
  int b  = k & 1;
  int kh = ph ? (a ? 2 : 0) : (a ? 3 : 1);
  int kw = pw ? (b ? 2 : 0) : (b ? 3 : 1);
  float v = w[((ci * COUT + co) * 4 + kh) * 4 + kw];
  Ah[idx] = (_Float16)v;
}

// ---------------------------------------------------------------------------
// Main: block = (2 output rows) x (128 Cout) x (64 ow') of one parity class.
// 8 waves: wave = (cg = co half, j = 32-pixel quarter). Each wave: 4 M-tiles
// x 2 N-tiles (8 v8f accumulators), so each B fragment feeds 4 WMMAs
// (0.5 ds_load_b128 per WMMA) and each A fragment feeds 2 WMMAs.
// ---------------------------------------------------------------------------
__global__ __launch_bounds__(256)
void convt2d_wmma_kernel(const float* __restrict__ x,
                         const _Float16* __restrict__ Ah,
                         const float* __restrict__ bias,
                         float* __restrict__ out) {
  const int g    = blockIdx.x;       // 0..31  row pair (rows 2g, 2g+1 of class)
  const int n    = blockIdx.y;       // 0..15  batch
  const int quad = blockIdx.z;       // 0..3   (ph,pw)
  const int ph = quad >> 1, pw = quad & 1;

  const int t    = threadIdx.x;
  const int lane = t & 31;
  const int wave = t >> 5;

  __shared__ float    xraw[96 * XROW_STRIDE];                          // ~26 KB
  __shared__ _Float16 bp[128 * BP_STRIDE] __attribute__((aligned(16))); // ~34 KB

  const int iwoff1 = pw ? 0 : -1;    // tap b=1 ; tap b=0 offset = iwoff1+1
  const int ihbase = 2 * g + (ph ? 0 : -1);   // xraw rel row 0..2 -> ih

  // ---- phase-2 constants: thread packs 16 pixels for one ci'
  const int ci2 = t & 31;
  const int pg2 = t >> 5;            // 0..7 -> pixel run [16*pg2, +16)

  // ---- compute constants (ISA fragment layouts)
  const int cg     = wave & 1;       // Cout half (64 co)
  const int nb0    = (wave >> 1) * 2;// first of 2 N-tiles
  const int lco    = lane & 15;      // A: M index; B/C/D: N index
  const int hi     = lane >> 4;      // half-wave selector
  // A: lane holds M=lco; K runs {0..7}&{16..23} (lo half) / {8..15}&{24..31}
  const _Float16* Abase =
      Ah + (size_t)((quad * COUT + cg * 64 + lco) * KTOT) + (hi << 3);

  v8f acc[8] = {};   // [m][e] : 4 co tiles x 2 pixel tiles, f32 accumulation

  for (int c = 0; c < NCHUNK; ++c) {
    // ------------- phase 1: stage 96 x-rows (3 ih x 32 ci) coalesced -------
#pragma unroll
    for (int i = 0; i < 6; ++i) {
      int flat = t + i * 256;              // 0..1535
      int r    = flat >> 4;                // 0..95 = rel*32 + ci'
      int q    = flat & 15;                // float4 index within row
      int rel  = r >> 5;
      int cir  = r & 31;
      int ih   = ihbase + rel;
      bool ok  = (ih >= 0) && (ih < HIN);
      const float4* src = (const float4*)(x +
          (((size_t)n * CIN + c * 32 + cir) * HIN + (ok ? ih : 0)) * WIN);
      float4 val = ok ? src[q] : make_float4(0.f, 0.f, 0.f, 0.f);
      ((float4*)(xraw + r * XROW_STRIDE))[q] = val;
    }
    __syncthreads();
    // ------------- phase 2: pack B panel [pixel][k'] f16 (rolling window) --
    {
      const int p0  = pg2 * 16;
      const int rp  = p0 >> 6;             // which of the 2 output rows
      const int ow0 = p0 & 63;
      const float* xl0 = xraw + ((rp + 1) * 32 + ci2) * XROW_STRIDE; // tap a=0
      const float* xl1 = xraw + ((rp    ) * 32 + ci2) * XROW_STRIDE; // tap a=1
      int ipv = ow0 + iwoff1;
      float pv0 = ((unsigned)ipv < (unsigned)WIN) ? xl0[ipv] : 0.f;
      float pv1 = ((unsigned)ipv < (unsigned)WIN) ? xl1[ipv] : 0.f;
#pragma unroll
      for (int j = 0; j < 16; ++j) {
        int ic = ow0 + j + iwoff1 + 1;     // tap b=0 column
        bool okc = (unsigned)ic < (unsigned)WIN;
        float c0 = okc ? xl0[ic] : 0.f;
        float c1 = okc ? xl1[ic] : 0.f;
        union { _Float16 h[4]; uint2 u; } pk;
        pk.h[0] = (_Float16)c0;  pk.h[1] = (_Float16)pv0;
        pk.h[2] = (_Float16)c1;  pk.h[3] = (_Float16)pv1;
        *(uint2*)(bp + (p0 + j) * BP_STRIDE + ci2 * 4) = pk.u;  // 8B aligned
        pv0 = c0; pv1 = c1;
      }
    }
    __syncthreads();
    // ------------- compute: 32 WMMAs per wave on this 128-K chunk ----------
    const _Float16* Ac = Abase + c * 128;
    if (c + 1 < NCHUNK) __builtin_prefetch(Ac + 128, 0, 0);  // global_prefetch_b8
#pragma unroll
    for (int kc = 0; kc < 128; kc += 32) {
      // B: lane holds N=lco; K runs {0..7}&{16..23} (lo half) / {8..15}&{24..31}
      v16h bf[2];
#pragma unroll
      for (int e = 0; e < 2; ++e) {
        const _Float16* bpp =
            bp + ((nb0 + e) * 16 + lco) * BP_STRIDE + kc + (hi << 3);
        v8h blo = *(const v8h*)bpp;
        v8h bhi = *(const v8h*)(bpp + 16);
        bf[e] = __builtin_shufflevector(blo, bhi,
            0, 1, 2, 3, 4, 5, 6, 7, 8, 9, 10, 11, 12, 13, 14, 15);
      }
#pragma unroll
      for (int m = 0; m < 4; ++m) {
        const _Float16* Ap = Ac + (size_t)m * 16 * KTOT + kc;
        v8h alo = *(const v8h*)Ap;
        v8h ahi = *(const v8h*)(Ap + 16);
        v16h af = __builtin_shufflevector(alo, ahi,
            0, 1, 2, 3, 4, 5, 6, 7, 8, 9, 10, 11, 12, 13, 14, 15);
        acc[m * 2 + 0] = __builtin_amdgcn_wmma_f32_16x16x32_f16(
            false, af, false, bf[0], (short)0, acc[m * 2 + 0], false, false);
        acc[m * 2 + 1] = __builtin_amdgcn_wmma_f32_16x16x32_f16(
            false, af, false, bf[1], (short)0, acc[m * 2 + 1], false, false);
      }
    }
    __syncthreads();
  }

  // ------------- epilogue: bias add + scatter per C/D layout ---------------
#pragma unroll
  for (int m = 0; m < 4; ++m) {
    float bco[8];
#pragma unroll
    for (int r = 0; r < 8; ++r)
      bco[r] = bias[cg * 64 + m * 16 + r + (hi << 3)];
#pragma unroll
    for (int e = 0; e < 2; ++e) {
      const int p   = (nb0 + e) * 16 + lco;
      const int rp  = p >> 6;
      const int oh  = 2 * (2 * g + rp) + ph;
      const int ow  = 2 * (p & 63) + pw;
#pragma unroll
      for (int r = 0; r < 8; ++r) {
        const int co = cg * 64 + m * 16 + r + (hi << 3);
        out[(((size_t)n * COUT + co) * HOUT + oh) * HOUT + ow] =
            acc[m * 2 + e][r] + bco[r];
      }
    }
  }
}

extern "C" void kernel_launch(void* const* d_in, const int* in_sizes, int n_in,
                              void* d_out, int out_size, void* d_ws, size_t ws_size,
                              hipStream_t stream) {
  const float* x    = (const float*)d_in[0];   // (16,256,64,64)
  const float* w    = (const float*)d_in[1];   // (256,128,4,4)
  const float* bias = (const float*)d_in[2];   // (128,)
  float* out = (float*)d_out;                  // (16,128,128,128)
  _Float16* Ah = (_Float16*)d_ws;              // 4*128*1024 f16 = 1 MB

  hipLaunchKernelGGL(pack_weights_kernel, dim3(2048), dim3(256), 0, stream, w, Ah);
  hipLaunchKernelGGL(convt2d_wmma_kernel, dim3(32, 16, 4), dim3(256), 0, stream,
                     x, Ah, bias, out);
}
